// GNNModel_86517821216405
// MI455X (gfx1250) — matrix operand order repaired and checked by reading, not compiled, not run on hardware
//
#include <hip/hip_runtime.h>

typedef __attribute__((ext_vector_type(16))) _Float16 v16h;
typedef __attribute__((ext_vector_type(8)))  float    v8f;

#define HID 64

// ---------------------------------------------------------------------------
// Zero two float regions (deterministic re-init each call)
// ---------------------------------------------------------------------------
__global__ void __launch_bounds__(256) zero2_kernel(float* a, int na, float* b, int nb) {
    int i = blockIdx.x * 256 + threadIdx.x;
    if (i < na)            a[i] = 0.0f;
    else if (i < na + nb)  b[i - na] = 0.0f;
}

// ---------------------------------------------------------------------------
// Node-level GEMM via WMMA:  Y[N,64] = act( rowscale(X[N,K]) @ Weff[K,64] + bias )
//   SUB : Weff = W0 - Wsub        (EdgeConv x_i term:  W_top - W_bot)
//   BIAS: add bias[n]
//   MEAN: rowscale = 1/max(cnt,1), relu output, cnt==0 rows forced to 0
// All flags are compile-time so the epilogue is branch-free. Weight tile is
// staged once per block in LDS pre-transposed to the B-fragment layout
// (WT[n][k], k padded to KP); each B fragment is two 16B LDS reads, 16 halves apart,
// preloaded into registers as a single clause. One wave -> one 16x64 tile.
// N rows must be a multiple of 16 (50000 = 3125*16).
// ---------------------------------------------------------------------------
template <int K, bool SUB, bool BIAS, bool MEAN>
__global__ void __launch_bounds__(256) node_gemm_kernel(
    const float* __restrict__ X,
    const float* __restrict__ W0, const float* __restrict__ Wsub,
    const float* __restrict__ bias, const float* __restrict__ cnt,
    float* __restrict__ Y, int n_tiles)
{
    constexpr int KP     = (K + 31) & ~31;      // 32 or 64
    constexpr int KSTEPS = KP / 32;             // 1 or 2
    __shared__ __align__(16) _Float16 WT[64 * KP];

    // ---- cooperative LDS fill: global reads coalesced over n, store transposed
    #pragma unroll
    for (int idx = threadIdx.x; idx < 64 * KP; idx += 256) {
        const int k = idx >> 6;                 // 0..KP-1
        const int n = idx & 63;
        float w = 0.0f;
        if (k < K) {
            w = W0[k * HID + n];
            if constexpr (SUB) w -= Wsub[k * HID + n];
        }
        WT[n * KP + k] = (_Float16)w;
    }
    __syncthreads();

    const int tile = blockIdx.x * 8 + (threadIdx.x >> 5);
    if (tile >= n_tiles) return;                // wave-uniform: EXEC all-1s for WMMA

    const int lane    = threadIdx.x & 31;
    const int half_id = lane >> 4;              // 0: lanes 0-15, 1: lanes 16-31
    const int col16   = lane & 15;
    const int m0      = tile * 16;
    const int arow    = m0 + col16;             // row this lane feeds into A fragment

    // ---- preload all B fragments (one ds clause, one wait)
    v16h bf[KSTEPS][4];
    #pragma unroll
    for (int ks = 0; ks < KSTEPS; ++ks)
        #pragma unroll
        for (int t = 0; t < 4; ++t)
            for (int s = 0; s < 8; ++s) {
                bf[ks][t][s]     = WT[(t * 16 + col16) * KP + ks * 32 + half_id * 8 + s];
                bf[ks][t][8 + s] = WT[(t * 16 + col16) * KP + ks * 32 + half_id * 8 + 16 + s];
            }

    // ---- mean factors: this lane's A-row scale + its 8 store-row cnt values
    float scale = 1.0f;
    float cv[8];
    if constexpr (MEAN) {
        scale = 1.0f / fmaxf(cnt[arow], 1.0f);
        const float4 c0 = *(const float4*)(cnt + m0 + half_id * 8);
        const float4 c1 = *(const float4*)(cnt + m0 + half_id * 8 + 4);
        cv[0] = c0.x; cv[1] = c0.y; cv[2] = c0.z; cv[3] = c0.w;
        cv[4] = c1.x; cv[5] = c1.y; cv[6] = c1.z; cv[7] = c1.w;
    }

    v8f acc[4] = {{}, {}, {}, {}};
    const float* xr = X + (size_t)arow * K;

    #pragma unroll
    for (int ks = 0; ks < KSTEPS; ++ks) {
        const int kb = ks * 32 + half_id * 8;
        // ---- A fragment (16x32 f16): halves 0..7 = K kb..kb+7, 8..15 = kb+16..kb+23
        v16h a;
        {
            float4 q0 = *(const float4*)(xr + kb);
            float4 q1 = *(const float4*)(xr + kb + 4);
            a[0] = (_Float16)(q0.x * scale); a[1] = (_Float16)(q0.y * scale);
            a[2] = (_Float16)(q0.z * scale); a[3] = (_Float16)(q0.w * scale);
            a[4] = (_Float16)(q1.x * scale); a[5] = (_Float16)(q1.y * scale);
            a[6] = (_Float16)(q1.z * scale); a[7] = (_Float16)(q1.w * scale);
            if constexpr (K >= 32) {
                float4 q2 = *(const float4*)(xr + kb + 16);
                float4 q3 = *(const float4*)(xr + kb + 20);
                a[8]  = (_Float16)(q2.x * scale); a[9]  = (_Float16)(q2.y * scale);
                a[10] = (_Float16)(q2.z * scale); a[11] = (_Float16)(q2.w * scale);
                a[12] = (_Float16)(q3.x * scale); a[13] = (_Float16)(q3.y * scale);
                a[14] = (_Float16)(q3.z * scale); a[15] = (_Float16)(q3.w * scale);
            } else {
                #pragma unroll
                for (int j = 8; j < 16; ++j) a[j] = (_Float16)0.0f;
            }
        }
        #pragma unroll
        for (int t = 0; t < 4; ++t)
            acc[t] = __builtin_amdgcn_wmma_f32_16x16x32_f16(
                false, a, false, bf[ks][t], (short)0, acc[t], false, false);
    }

    // ---- branch-free store: one base pointer, all-immediate offsets
    // D layout: VGPR r -> row m0 + half_id*8 + r, col = t*16 + col16
    float* yp = Y + (size_t)(m0 + half_id * 8) * HID + col16;
    #pragma unroll
    for (int t = 0; t < 4; ++t) {
        float bv = 0.0f;
        if constexpr (BIAS) bv = bias[t * 16 + col16];
        #pragma unroll
        for (int r = 0; r < 8; ++r) {
            float v = acc[t][r] + bv;
            if constexpr (MEAN) {
                v = fmaxf(v, 0.0f);
                if (cv[r] == 0.0f) v = 0.0f;   // empty segment -> 0 (reference)
            }
            yp[r * HID + t * 16] = v;
        }
    }
}

// ---------------------------------------------------------------------------
// Per-edge: S[dst] += relu(A[dst] + B[src]); one wave per edge, float2/lane.
// cnt accumulated only on the first layer (same dst both layers).
// ---------------------------------------------------------------------------
__global__ void __launch_bounds__(256) edge_accum_kernel(
    const int* __restrict__ src, const int* __restrict__ dst,
    const float* __restrict__ A, const float* __restrict__ B,
    float* __restrict__ S, float* __restrict__ cnt, int E, int do_cnt)
{
    int e    = (blockIdx.x * 256 + threadIdx.x) >> 5;
    int lane = threadIdx.x & 31;
    if (e >= E) return;
    const int s = src[e];
    const int d = dst[e];
    const float2 av = *(const float2*)(A + (size_t)d * HID + lane * 2);
    const float2 bv = *(const float2*)(B + (size_t)s * HID + lane * 2);
    float v0 = fmaxf(av.x + bv.x, 0.0f);
    float v1 = fmaxf(av.y + bv.y, 0.0f);
    atomicAdd(S + (size_t)d * HID + lane * 2 + 0, v0);
    atomicAdd(S + (size_t)d * HID + lane * 2 + 1, v1);
    if (do_cnt && lane == 0) atomicAdd(cnt + d, 1.0f);
}

// ---------------------------------------------------------------------------
// out[N,2] = H[N,64] @ wl[64,2] + bl  (tiny head; one thread per node)
// ---------------------------------------------------------------------------
__global__ void __launch_bounds__(256) final_linear_kernel(
    const float* __restrict__ H, const float* __restrict__ wl,
    const float* __restrict__ bl, float* __restrict__ out, int N)
{
    int n = blockIdx.x * 256 + threadIdx.x;
    if (n >= N) return;
    float acc0 = bl[0], acc1 = bl[1];
    const float* h = H + (size_t)n * HID;
    #pragma unroll 8
    for (int k = 0; k < HID; ++k) {
        float hv = h[k];
        acc0 = fmaf(hv, wl[k * 2 + 0], acc0);
        acc1 = fmaf(hv, wl[k * 2 + 1], acc1);
    }
    out[n * 2 + 0] = acc0;
    out[n * 2 + 1] = acc1;
}

extern "C" void kernel_launch(void* const* d_in, const int* in_sizes, int n_in,
                              void* d_out, int out_size, void* d_ws, size_t ws_size,
                              hipStream_t stream) {
    const float* x   = (const float*)d_in[0];
    const int*   ei  = (const int*)  d_in[1];
    const float* w1a = (const float*)d_in[2];
    const float* b1a = (const float*)d_in[3];
    const float* w1b = (const float*)d_in[4];
    const float* b1b = (const float*)d_in[5];
    const float* w2a = (const float*)d_in[6];
    const float* b2a = (const float*)d_in[7];
    const float* w2b = (const float*)d_in[8];
    const float* b2b = (const float*)d_in[9];
    const float* wl  = (const float*)d_in[10];
    const float* bl  = (const float*)d_in[11];
    float* out = (float*)d_out;

    const int N = in_sizes[0] / 16;   // 50000
    const int E = in_sizes[1] / 2;    // 800000
    const int* src = ei;
    const int* dst = ei + E;

    // workspace layout (floats): A | B | S | H | cnt   (~51.5 MB)
    float* ws  = (float*)d_ws;
    float* A   = ws;
    float* B   = A + (size_t)N * HID;
    float* S   = B + (size_t)N * HID;
    float* H   = S + (size_t)N * HID;
    float* cnt = H + (size_t)N * HID;

    const int n_tiles     = N / 16;                         // 3125 (exact)
    const int gemm_blocks = (n_tiles + 7) / 8;              // 8 waves/block
    const int edge_blocks = (E + 7) / 8;
    const int zero_all    = (int)(((size_t)N * HID + N + 255) / 256);
    const int zero_s      = (int)(((size_t)N * HID + 255) / 256);
    dim3 blk(256);

    // ---- Layer 1 ----
    zero2_kernel<<<zero_all, blk, 0, stream>>>(S, N * HID, cnt, N);
    // A1 = x @ (W1_top - W1_bot) + b1a ; B1 = x @ W1_bot     (K = 16, zero-padded to 32)
    node_gemm_kernel<16, true, true, false><<<gemm_blocks, blk, 0, stream>>>(
        x, w1a, w1a + 16 * HID, b1a, nullptr, A, n_tiles);
    node_gemm_kernel<16, false, false, false><<<gemm_blocks, blk, 0, stream>>>(
        x, w1a + 16 * HID, nullptr, nullptr, nullptr, B, n_tiles);
    edge_accum_kernel<<<edge_blocks, blk, 0, stream>>>(src, dst, A, B, S, cnt, E, 1);
    // h1 = relu( (S / max(cnt,1)) @ w1b + b1b ), empty segments -> 0
    node_gemm_kernel<64, false, true, true><<<gemm_blocks, blk, 0, stream>>>(
        S, w1b, nullptr, b1b, cnt, H, n_tiles);

    // ---- Layer 2 ----
    zero2_kernel<<<zero_s, blk, 0, stream>>>(S, N * HID, S, 0);
    node_gemm_kernel<64, true, true, false><<<gemm_blocks, blk, 0, stream>>>(
        H, w2a, w2a + 64 * HID, b2a, nullptr, A, n_tiles);
    node_gemm_kernel<64, false, false, false><<<gemm_blocks, blk, 0, stream>>>(
        H, w2a + 64 * HID, nullptr, nullptr, nullptr, B, n_tiles);
    edge_accum_kernel<<<edge_blocks, blk, 0, stream>>>(src, dst, A, B, S, cnt, E, 0);
    node_gemm_kernel<64, false, true, true><<<gemm_blocks, blk, 0, stream>>>(
        S, w2b, nullptr, b2b, cnt, H, n_tiles);

    // ---- Output head ----
    final_linear_kernel<<<(N + 255) / 256, blk, 0, stream>>>(H, wl, bl, out, N);
}
